// GenSP_43636867728097
// MI455X (gfx1250) — matrix operand order repaired and checked by reading, not compiled, not run on hardware
//
#include <hip/hip_runtime.h>
#include <hip/hip_bf16.h>

typedef __attribute__((ext_vector_type(16))) __bf16 v16bf;
typedef __attribute__((ext_vector_type(8)))  float  v8f;
typedef __attribute__((ext_vector_type(4)))  unsigned int u32x4;
typedef __attribute__((ext_vector_type(8)))  int    i32x8;
typedef __attribute__((ext_vector_type(4)))  int    i32x4;

#define CCH 64          // channels
#define HW  384         // image H == W
#define SHW 16          // superpixel block side (stoken)
#define NS  24          // superpixels per side
#define NSP 576         // total superpixels
#define NPX (384*384)   // total pixels
#define CSTR 65         // candidate LDS row stride (TDM pad: +1 DW per 64 DW)

// ---------------------------------------------------------------------------
// K1: initial centroids = 16x16 block means.  1 block per superpixel, 8 waves,
// each wave owns 8 channels; wave-reduce with __shfl_xor (wave32).
// ---------------------------------------------------------------------------
__global__ __launch_bounds__(256) void k_init_cent(const float* __restrict__ x,
                                                   float* __restrict__ cent) {
  const int t = threadIdx.x, lane = t & 31, wave = t >> 5;
  const int s = blockIdx.x, sy = s / NS, sx = s % NS;
  for (int c8 = 0; c8 < 8; ++c8) {
    const int c = wave * 8 + c8;
    float sum = 0.f;
#pragma unroll
    for (int j = 0; j < 8; ++j) {
      const int m = lane + 32 * j;                     // pixel-in-block
      sum += x[c * NPX + (sy * SHW + (m >> 4)) * HW + sx * SHW + (m & 15)];
    }
#pragma unroll
    for (int o = 1; o < 32; o <<= 1) sum += __shfl_xor(sum, o, 32);
    if (lane == 0) cent[s * CCH + c] = sum * (1.f / 256.f);
  }
}

// ---------------------------------------------------------------------------
// K3/K5: one SSN iteration for one superpixel block.
//   candidate centroids : TDM gather-load (tensor_load_to_lds, 9 indexed rows)
//   dist GEMM : D(256x16) = Pixels(256x64) x Cand^T(64x16)  via v_wmma bf16
//   softmax over 16-lane candidate dim (shfl_xor width 16)
//   LAST=0 : update GEMM num(16x64) = Aff^T(16x256) x Pixels(256x64) via wmma
//   LAST=1 : argmax -> klabel
// ---------------------------------------------------------------------------
template <bool LAST>
__global__ __launch_bounds__(256) void k_iter(const float* __restrict__ x,
                                              const float* __restrict__ cent,
                                              float* __restrict__ num,
                                              float* __restrict__ den,
                                              int* __restrict__ klabel) {
  __shared__ __bf16 sPx[256 * 72];     // pixel features, row-padded to 72
  __shared__ float  sCandF[16 * CSTR]; // candidate centroids f32 (TDM dest)
  __shared__ float  sAff[256 * 16];    // affinities (update path)
  __shared__ float  sNum[16 * 64];     // per-block centroid numerators
  __shared__ float  sPn[256];          // pixel squared norms (bf16-rounded)
  __shared__ float  sCn[16];           // candidate squared norms
  __shared__ float  sDen[16];          // per-block denominators

  const int t = threadIdx.x, lane = t & 31, wave = t >> 5;
  const int s = blockIdx.x, sy = s / NS, sx = s % NS;

  // ---- TDM gather: fetch the 9 candidate centroid rows (64 f32 each) from
  //      cent[] into sCandF, one row per candidate slot, LDS stride 65 DW via
  //      descriptor padding (pad_interval=64DW, pad_amount=1DW).
  if (wave == 0) {
    int rowidx[9];
#pragma unroll
    for (int k = 0; k < 9; ++k) {
      const int cy = sy + k / 3 - 1, cx = sx + k % 3 - 1;
      rowidx[k] = (cy >= 0 && cy < NS && cx >= 0 && cx < NS) ? (cy * NS + cx) : 0;
    }
    const unsigned ldsAddr = (unsigned)(size_t)(void*)sCandF;   // LDS byte offset
    const unsigned long long ga = (unsigned long long)(size_t)cent;
    u32x4 g0; i32x8 g1; i32x4 g2, g3; i32x8 g4;
    g0[0] = 0x80000001u;                               // count=1, gather_mode, 16b idx
    g0[1] = ldsAddr;                                   // lds_addr
    g0[2] = (unsigned)(ga & 0xFFFFFFFFu);              // global_addr[31:0]
    g0[3] = (unsigned)((ga >> 32) & 0x01FFFFFFu) | (2u << 30);  // addr[56:32]|type=2
    g1[0] = (2 << 16) | (1 << 20) | (5 << 22);         // data_size=4B, pad_en, ivl=64DW
    g1[1] = (CCH & 0xFFFF) << 16;                      // tensor_dim0 = 64
    g1[2] = (NSP & 0xFFFF) << 16;                      // tensor_dim1 = 576 rows
    g1[3] = (CCH << 16);                               // tile_dim0 = 64
    g1[4] = 9;                                         // tile_dim1 = 9 gather rows
    g1[5] = CCH;                                       // tensor_dim0_stride = 64
    g1[6] = 0; g1[7] = 0;
    g2[0] = rowidx[0] | (rowidx[1] << 16);
    g2[1] = rowidx[2] | (rowidx[3] << 16);
    g2[2] = rowidx[4] | (rowidx[5] << 16);
    g2[3] = rowidx[6] | (rowidx[7] << 16);
    g3[0] = rowidx[8]; g3[1] = 0; g3[2] = 0; g3[3] = 0;
    g4[0] = 0; g4[1] = 0; g4[2] = 0; g4[3] = 0;
    g4[4] = 0; g4[5] = 0; g4[6] = 0; g4[7] = 0;
    __builtin_amdgcn_tensor_load_to_lds(g0, g1, g2, g3, g4, 0);
  }

  // ---- load pixels (overlaps the TDM DMA): thread t <-> block pixel t
  {
    const int row = t >> 4, col = t & 15;
    const float* px = x + (sy * SHW + row) * HW + sx * SHW + col;
    float nrm = 0.f;
#pragma unroll
    for (int c = 0; c < CCH; ++c) {
      const __bf16 vb = (__bf16)px[c * NPX];
      sPx[t * 72 + c] = vb;
      const float vf = (float)vb;
      nrm += vf * vf;
    }
    sPn[t] = nrm;
  }

  if (wave == 0) __builtin_amdgcn_s_wait_tensorcnt(0);
  __syncthreads();

  // ---- zero invalid candidate rows (TDM fetched clamped index 0) and the
  //      padded rows 9..15, then candidate norms of the bf16-rounded values.
  for (int idx = t; idx < 16 * CCH; idx += 256) {
    const int nn = idx >> 6, c = idx & 63;
    bool v = false;
    if (nn < 9) {
      const int cy = sy + nn / 3 - 1, cx = sx + nn % 3 - 1;
      v = (cy >= 0 && cy < NS && cx >= 0 && cx < NS);
    }
    if (!v) sCandF[nn * CSTR + c] = 0.f;
  }
  if (t < 16) sDen[t] = 0.f;
  __syncthreads();
  if (t < 16) {
    float nn = 0.f;
    for (int c = 0; c < CCH; ++c) {
      const float vf = (float)(__bf16)sCandF[t * CSTR + c];
      nn += vf * vf;
    }
    sCn[t] = nn;
  }
  __syncthreads();

  const int n  = lane & 15;          // candidate index / N column
  const int hi = (lane >> 4) & 1;    // half-wave select
  bool nvalid = false;
  if (n < 9) {
    const int cy = sy + n / 3 - 1, cx = sx + n % 3 - 1;
    nvalid = (cy >= 0 && cy < NS && cx >= 0 && cx < NS);
  }
  const float cn = sCn[n];

  float dsum = 0.f;
#pragma unroll
  for (int mt2 = 0; mt2 < 2; ++mt2) {
    const int mt = wave * 2 + mt2;                 // 16-row pixel tile
    v8f acc = {0.f, 0.f, 0.f, 0.f, 0.f, 0.f, 0.f, 0.f};
#pragma unroll
    for (int kb = 0; kb < CCH; kb += 32) {
      v16bf a, b;
      // A (16-bit 16x32): row = mt*16 + (lane&15); lane<16 -> K {0..7,16..23}
      const __bf16* ap = &sPx[(mt * 16 + n) * 72 + kb + hi * 8];
#pragma unroll
      for (int j = 0; j < 8; ++j) { a[j] = ap[j]; a[j + 8] = ap[j + 16]; }
      // B (16-bit 32x16): N = lane&15, lane<16 -> K {0..7,16..23}
      const float* bp = &sCandF[n * CSTR + kb + hi * 8];
#pragma unroll
      for (int j = 0; j < 8; ++j) { b[j] = (__bf16)bp[j]; b[j + 8] = (__bf16)bp[j + 16]; }
      acc = __builtin_amdgcn_wmma_f32_16x16x32_bf16(false, a, false, b,
                                                    (short)0, acc, false, false);
    }
    // D layout: VGPR r, lane: M = r + 8*hi, N = lane&15
#pragma unroll
    for (int r = 0; r < 8; ++r) {
      const int m = mt * 16 + r + hi * 8;          // pixel in block
      const float dist  = sPn[m] + cn - 2.f * acc[r];
      const float logit = nvalid ? -dist : -__builtin_inff();
      float mx = logit;
#pragma unroll
      for (int o = 1; o < 16; o <<= 1) mx = fmaxf(mx, __shfl_xor(mx, o, 16));
      const float e = __expf(logit - mx);
      float ssum = e;
#pragma unroll
      for (int o = 1; o < 16; o <<= 1) ssum += __shfl_xor(ssum, o, 16);
      const float aff = e / ssum;                  // 0 for invalid candidates
      if (LAST) {
        float bv = logit; int bi = n;              // argmax, lowest-index ties
#pragma unroll
        for (int o = 1; o < 16; o <<= 1) {
          const float ov = __shfl_xor(bv, o, 16);
          const int   oi = __shfl_xor(bi, o, 16);
          if (ov > bv || (ov == bv && oi < bi)) { bv = ov; bi = oi; }
        }
        if (n == 0)
          klabel[(sy * SHW + (m >> 4)) * HW + sx * SHW + (m & 15)] = bi;
      } else {
        sAff[m * 16 + n] = aff;
        dsum += aff;
      }
    }
  }

  if (!LAST) {
    dsum += __shfl_xor(dsum, 16, 32);              // fold the two half-waves
    if (lane < 16 && nvalid) atomicAdd(&sDen[n], dsum);
    __syncthreads();
    // ---- update GEMM: waves 0..3, one 16-channel tile each, K = 256 pixels
    if (wave < 4) {
      const int ct = wave;
      v8f acc2 = {0.f, 0.f, 0.f, 0.f, 0.f, 0.f, 0.f, 0.f};
#pragma unroll
      for (int kc = 0; kc < 8; ++kc) {
        const int pb = kc * 32;
        v16bf a, b;
        const int aoff = pb + hi * 8;              // A: M=cand, K=pixel
#pragma unroll
        for (int j = 0; j < 8; ++j) {
          a[j]     = (__bf16)sAff[(aoff + j) * 16 + n];
          a[j + 8] = (__bf16)sAff[(aoff + 16 + j) * 16 + n];
        }
        const int boff = pb + hi * 8;              // B: K=pixel, N=channel
#pragma unroll
        for (int j = 0; j < 8; ++j) {
          b[j]     = sPx[(boff + j) * 72 + ct * 16 + n];
          b[j + 8] = sPx[(boff + 16 + j) * 72 + ct * 16 + n];
        }
        acc2 = __builtin_amdgcn_wmma_f32_16x16x32_bf16(false, a, false, b,
                                                       (short)0, acc2, false, false);
      }
#pragma unroll
      for (int r = 0; r < 8; ++r)
        sNum[(r + hi * 8) * 64 + ct * 16 + n] = acc2[r];
    }
    __syncthreads();
    // ---- scatter per-block partials into global num/den (few atomics)
    for (int idx = t; idx < 9 * 64 + 9; idx += 256) {
      const int nn = (idx < 9 * 64) ? (idx >> 6) : (idx - 9 * 64);
      const int cy = sy + nn / 3 - 1, cx = sx + nn % 3 - 1;
      if (cy >= 0 && cy < NS && cx >= 0 && cx < NS) {
        if (idx < 9 * 64)
          atomicAdd(&num[(cy * NS + cx) * CCH + (idx & 63)], sNum[nn * 64 + (idx & 63)]);
        else
          atomicAdd(&den[cy * NS + cx], sDen[nn]);
      }
    }
  }
}

// ---------------------------------------------------------------------------
// K4: cent = num / (den + 1e-16)   (mode 0, SSN update)
//     cent = num / max(den, 1)     (mode 1, painting means)
// ---------------------------------------------------------------------------
__global__ void k_norm(const float* __restrict__ num, const float* __restrict__ den,
                       float* __restrict__ cent, int mode) {
  const int i = blockIdx.x * blockDim.x + threadIdx.x;
  if (i >= NSP * CCH) return;
  const float d = den[i >> 6];
  cent[i] = num[i] / (mode ? fmaxf(d, 1.f) : (d + 1e-16f));
}

// ---------------------------------------------------------------------------
// K6a: per-superpixel sums/counts of assigned pixels (hard labels).
// ---------------------------------------------------------------------------
__global__ __launch_bounds__(256) void k_paint_acc(const float* __restrict__ x,
                                                   const int* __restrict__ klabel,
                                                   float* __restrict__ sums,
                                                   float* __restrict__ cnt) {
  __shared__ float s9[9 * 64];
  __shared__ float c9[9];
  const int t = threadIdx.x;
  const int s = blockIdx.x, sy = s / NS, sx = s % NS;
  for (int i = t; i < 9 * 64; i += 256) s9[i] = 0.f;
  if (t < 9) c9[t] = 0.f;
  __syncthreads();
  const int gp = (sy * SHW + (t >> 4)) * HW + sx * SHW + (t & 15);
  const int k = klabel[gp];
  atomicAdd(&c9[k], 1.f);
  for (int c = 0; c < CCH; ++c) atomicAdd(&s9[k * 64 + c], x[c * NPX + gp]);
  __syncthreads();
  for (int idx = t; idx < 9 * 64 + 9; idx += 256) {
    const int nn = (idx < 9 * 64) ? (idx >> 6) : (idx - 9 * 64);
    const int cy = sy + nn / 3 - 1, cx = sx + nn % 3 - 1;
    if (cy >= 0 && cy < NS && cx >= 0 && cx < NS) {
      if (idx < 9 * 64)
        atomicAdd(&sums[(cy * NS + cx) * CCH + (idx & 63)], s9[nn * 64 + (idx & 63)]);
      else
        atomicAdd(&cnt[cy * NS + cx], c9[nn]);
    }
  }
}

// ---------------------------------------------------------------------------
// K6c: paint means back: out[c][p] = means[label(p)][c]
// ---------------------------------------------------------------------------
__global__ void k_paint(const float* __restrict__ means, const int* __restrict__ klabel,
                        float* __restrict__ out) {
  const int i = blockIdx.x * blockDim.x + threadIdx.x;
  if (i >= CCH * NPX) return;
  const int c = i / NPX, p = i % NPX;
  const int sy = (p / HW) / SHW, sx = (p % HW) / SHW;
  const int k = klabel[p];
  const int cy = sy + k / 3 - 1, cx = sx + k % 3 - 1;
  out[i] = means[(cy * NS + cx) * CCH + c];
}

// ---------------------------------------------------------------------------
extern "C" void kernel_launch(void* const* d_in, const int* in_sizes, int n_in,
                              void* d_out, int out_size, void* d_ws, size_t ws_size,
                              hipStream_t stream) {
  (void)in_sizes; (void)n_in; (void)out_size; (void)ws_size;
  const float* x = (const float*)d_in[0];   // (1,64,384,384) fp32; d_in[1]=stoken=16
  float* out = (float*)d_out;

  char* ws = (char*)d_ws;                   // 887,040 bytes used
  float* cent   = (float*)(ws);             // 576*64 f32
  float* num    = (float*)(ws + 147456);    // 576*64 f32
  float* den    = (float*)(ws + 294912);    // 576   f32 (contiguous after num)
  int*   klabel = (int*)  (ws + 297216);    // 147456 i32

  k_init_cent<<<NSP, 256, 0, stream>>>(x, cent);
  for (int it = 0; it < 2; ++it) {          // N_ITER-1 centroid updates
    (void)hipMemsetAsync(num, 0, (NSP * CCH + NSP) * sizeof(float), stream);
    k_iter<false><<<NSP, 256, 0, stream>>>(x, cent, num, den, klabel);
    k_norm<<<(NSP * CCH + 255) / 256, 256, 0, stream>>>(num, den, cent, 0);
  }
  k_iter<true><<<NSP, 256, 0, stream>>>(x, cent, num, den, klabel);

  (void)hipMemsetAsync(num, 0, (NSP * CCH + NSP) * sizeof(float), stream);
  k_paint_acc<<<NSP, 256, 0, stream>>>(x, klabel, num, den);
  k_norm<<<(NSP * CCH + 255) / 256, 256, 0, stream>>>(num, den, cent, 1);
  k_paint<<<(CCH * NPX + 255) / 256, 256, 0, stream>>>(cent, klabel, out);
}
